// DiagonalProjectionComposition_2078764171786
// MI455X (gfx1250) — compile-verified, matrix operand order checked by reading, not run on hardware
//
#include <hip/hip_runtime.h>

// out = A @ W2 @ B.T, W2 = Gaussian band (half-width 54, w2(d)=exp(-d^2/64)).
// Phase 1: T = A @ W2 as a per-row 109-tap convolution, emitting f16 hi/lo
//          split planes directly (f32 T never materialized).
// Phase 2: split B into f16 hi/lo planes (one-time, instead of per-tile-use).
// Phase 3: out = T @ B^T as a compute-bound (AI ~860 FLOP/B) split-precision
//          GEMM: 3x v_wmma_f32_16x16x32_f16 per K-step (hi*hi + lo*hi + hi*lo
//          ~ f32 accuracy), double-buffered LDS filled by async global->LDS
//          copies (ASYNCcnt) so the fill overlaps the WMMA stream.

typedef __attribute__((ext_vector_type(16))) _Float16 v16h;
typedef __attribute__((ext_vector_type(8)))  _Float16 v8h;
typedef __attribute__((ext_vector_type(4)))  _Float16 v4h;
typedef __attribute__((ext_vector_type(8)))  float    v8f;
typedef __attribute__((ext_vector_type(4)))  int      v4i;

#define M_DIM 8192
#define N_DIM 4096
#define HW    54
#define TAPS  (2 * HW + 1)

#if defined(__has_builtin)
#if __has_builtin(__builtin_amdgcn_global_load_async_to_lds_b128)
#define HAVE_ASYNC_LDS 1
#endif
#endif
#ifndef HAVE_ASYNC_LDS
#define HAVE_ASYNC_LDS 0
#endif

__device__ __forceinline__ void split_f32(float x, _Float16& h, _Float16& l) {
    _Float16 hh = (_Float16)x;
    h = hh;
    l = (_Float16)(x - (float)hh);
}

// 16-byte global -> LDS copy: async (no VGPR round trip, ASYNCcnt) if available.
// Builtin signature (from hipcc diagnostic): param0 = v4i addrspace(1)*,
// non-const; param1 = LDS destination; params 2/3 = imm offset, cpol.
__device__ __forceinline__ void cp16_g2l(const _Float16* g, _Float16* l) {
#if HAVE_ASYNC_LDS
    _Float16* gnc = const_cast<_Float16*>(g);
    __builtin_amdgcn_global_load_async_to_lds_b128(
        (__attribute__((address_space(1))) v4i*)gnc,
        (__attribute__((address_space(3))) v4i*)l, 0, 0);
#else
    *(v8h*)l = *(const v8h*)g;
#endif
}

__device__ __forceinline__ void async_fence() {
#if HAVE_ASYNC_LDS
#if __has_builtin(__builtin_amdgcn_s_wait_asynccnt)
    __builtin_amdgcn_s_wait_asynccnt(0);
#else
    asm volatile("s_wait_asynccnt 0x0" ::: "memory");
#endif
#endif
}

// ---------------------------------------------------------------------------
// Phase 1: T[i,j] = sum_{|d|<=54} A[i,j+d]*exp(-d^2/64), written as f16 hi/lo.
// ---------------------------------------------------------------------------
__global__ __launch_bounds__(256) void band_conv_split_kernel(const float* __restrict__ A,
                                                              _Float16* __restrict__ Thi,
                                                              _Float16* __restrict__ Tlo) {
    __shared__ float w2s[TAPS];
    __shared__ float seg[256 + 2 * HW];

    const int tid = threadIdx.x;
    const int row = blockIdx.y;
    const int j0  = blockIdx.x * 256;

    if (tid < TAPS) {
        float d = (float)(tid - HW);
        w2s[tid] = expf(-(d * d) * (1.0f / 64.0f));   // W1^2 = exp(-d^2/64)
    }
    const float* Arow = A + (size_t)row * N_DIM;
    for (int s = tid; s < 256 + 2 * HW; s += 256) {
        int k = j0 - HW + s;
        seg[s] = (k >= 0 && k < N_DIM) ? Arow[k] : 0.0f;
    }
    __syncthreads();

    float acc = 0.0f;
#pragma unroll 8
    for (int t = 0; t < TAPS; ++t)
        acc = fmaf(seg[tid + t], w2s[t], acc);

    _Float16 h, l;
    split_f32(acc, h, l);
    const size_t o = (size_t)row * N_DIM + j0 + tid;
    Thi[o] = h;
    Tlo[o] = l;
}

// ---------------------------------------------------------------------------
// Phase 2: one-time f32 -> (hi,lo) f16 split of B.
// ---------------------------------------------------------------------------
__global__ __launch_bounds__(256) void split_b_kernel(const float* __restrict__ B,
                                                      _Float16* __restrict__ Bhi,
                                                      _Float16* __restrict__ Blo) {
    const size_t i4 = (size_t)blockIdx.x * 256 + threadIdx.x;  // float4 index
    float4 v = ((const float4*)B)[i4];
    _Float16 h0, h1, h2, h3, l0, l1, l2, l3;
    split_f32(v.x, h0, l0);
    split_f32(v.y, h1, l1);
    split_f32(v.z, h2, l2);
    split_f32(v.w, h3, l3);
    v4h h = {h0, h1, h2, h3};
    v4h l = {l0, l1, l2, l3};
    ((v4h*)Bhi)[i4] = h;
    ((v4h*)Blo)[i4] = l;
}

// ---------------------------------------------------------------------------
// Phase 3: out = T @ B^T, split-precision WMMA GEMM.
// Block tile 128x128, K-step 32, 8 waves each owning a 32x64 C tile
// (2x4 WMMA accumulators -> 24 wmma per wave per K-step vs 12 frag loads).
// NT layout: A-frag (rows of T) and B-frag (rows of B = cols of B^T) share the
// same per-lane pattern: row = lane&15, 8-half chunks at (lane>>4)*8 and +16.
// ---------------------------------------------------------------------------
#define BM    128
#define BN    128
#define BK    32
#define LDW   40                 // padded LDS row stride in halves (80 B)
#define PLANE (BM * LDW)         // halves per LDS plane

__device__ __forceinline__ v16h load_frag(const _Float16* tile, int row0, int lane) {
    const int r  = row0 + (lane & 15);
    const int c0 = (lane >> 4) << 3;                 // 0 or 8 halves
    const _Float16* p = tile + r * LDW + c0;
    v8h lo = *(const v8h*)(p);                       // K = c0 .. c0+7
    v8h hi = *(const v8h*)(p + 16);                  // K = c0+16 .. c0+23
    return __builtin_shufflevector(lo, hi, 0, 1, 2, 3, 4, 5, 6, 7,
                                           8, 9, 10, 11, 12, 13, 14, 15);
}

// Each of 256 threads moves one 32B row-chunk per plane: row = tid>>1,
// half-offset = (tid&1)*16 -> 8 x 16B async copies per thread per K-step.
__device__ __forceinline__ void fill_planes(const _Float16* Ah, const _Float16* Al,
                                            const _Float16* Bh, const _Float16* Bl,
                                            _Float16* L, int kb, int tid) {
    const int r    = tid >> 1;
    const int koff = (tid & 1) * 16;
    const size_t go = (size_t)r * N_DIM + kb + koff;
    const int    lo = r * LDW + koff;
    cp16_g2l(Ah + go, L + 0 * PLANE + lo);  cp16_g2l(Ah + go + 8, L + 0 * PLANE + lo + 8);
    cp16_g2l(Al + go, L + 1 * PLANE + lo);  cp16_g2l(Al + go + 8, L + 1 * PLANE + lo + 8);
    cp16_g2l(Bh + go, L + 2 * PLANE + lo);  cp16_g2l(Bh + go + 8, L + 2 * PLANE + lo + 8);
    cp16_g2l(Bl + go, L + 3 * PLANE + lo);  cp16_g2l(Bl + go + 8, L + 3 * PLANE + lo + 8);
}

__global__ __launch_bounds__(256) void gemm_tbt_f16x3_kernel(const _Float16* __restrict__ Thi,
                                                             const _Float16* __restrict__ Tlo,
                                                             const _Float16* __restrict__ Bhi,
                                                             const _Float16* __restrict__ Blo,
                                                             float* __restrict__ C) {
    __shared__ __align__(16) _Float16 lds[2][4 * PLANE];   // [buf][Ahi,Alo,Bhi,Blo]

    const int tid  = threadIdx.x;
    const int lane = tid & 31;
    const int wid  = tid >> 5;          // 0..7
    const int m0   = (wid & 3) * 32;    // 4 waves along M
    const int n0   = (wid >> 2) * 64;   // 2 waves along N

    const size_t aOff = (size_t)(blockIdx.y * BM) * N_DIM;
    const size_t bOff = (size_t)(blockIdx.x * BN) * N_DIM;
    const _Float16* Ah = Thi + aOff;
    const _Float16* Al = Tlo + aOff;
    const _Float16* Bh = Bhi + bOff;
    const _Float16* Bl = Blo + bOff;

    v8f acc[2][4] = {};

    int buf = 0;
    fill_planes(Ah, Al, Bh, Bl, lds[0], 0, tid);
    async_fence();
    __syncthreads();

    for (int kb = 0; kb < N_DIM; kb += BK) {
        // kick off next tile's async fills; they run under the WMMA stream
        if (kb + BK < N_DIM)
            fill_planes(Ah, Al, Bh, Bl, lds[buf ^ 1], kb + BK, tid);

        const _Float16* L = lds[buf];
        v16h aH[2], aL[2], bH[4], bL[4];
#pragma unroll
        for (int i = 0; i < 2; ++i) {
            aH[i] = load_frag(L + 0 * PLANE, m0 + i * 16, lane);
            aL[i] = load_frag(L + 1 * PLANE, m0 + i * 16, lane);
        }
#pragma unroll
        for (int j = 0; j < 4; ++j) {
            bH[j] = load_frag(L + 2 * PLANE, n0 + j * 16, lane);
            bL[j] = load_frag(L + 3 * PLANE, n0 + j * 16, lane);
        }
#pragma unroll
        for (int i = 0; i < 2; ++i)
#pragma unroll
            for (int j = 0; j < 4; ++j) {
                acc[i][j] = __builtin_amdgcn_wmma_f32_16x16x32_f16(
                    false, aH[i], false, bH[j], (short)0, acc[i][j], false, false);
                acc[i][j] = __builtin_amdgcn_wmma_f32_16x16x32_f16(
                    false, aL[i], false, bH[j], (short)0, acc[i][j], false, false);
                acc[i][j] = __builtin_amdgcn_wmma_f32_16x16x32_f16(
                    false, aH[i], false, bL[j], (short)0, acc[i][j], false, false);
            }

        async_fence();       // next buffer's fills landed
        __syncthreads();     // everyone done reading current buffer
        buf ^= 1;
    }

    // C/D frag layout: element (M = r + (lane>=16)*8, N = lane&15) in VGPR r
#pragma unroll
    for (int i = 0; i < 2; ++i) {
#pragma unroll
        for (int j = 0; j < 4; ++j) {
            const int rowBase = blockIdx.y * BM + m0 + i * 16 + ((lane >> 4) << 3);
            const int col     = blockIdx.x * BN + n0 + j * 16 + (lane & 15);
#pragma unroll
            for (int r = 0; r < 8; ++r)
                C[(size_t)(rowBase + r) * N_DIM + col] = acc[i][j][r];
        }
    }
}

// ---------------------------------------------------------------------------
extern "C" void kernel_launch(void* const* d_in, const int* in_sizes, int n_in,
                              void* d_out, int out_size, void* d_ws, size_t ws_size,
                              hipStream_t stream) {
    (void)in_sizes; (void)n_in; (void)out_size; (void)ws_size;
    const float* A = (const float*)d_in[0];   // 8192 x 4096 f32
    const float* B = (const float*)d_in[1];   // 4096 x 4096 f32
    float* out = (float*)d_out;               // 8192 x 4096 f32

    // workspace: T hi/lo (64 MB each) + B hi/lo (32 MB each) = 192 MB
    _Float16* Thi = (_Float16*)d_ws;
    _Float16* Tlo = Thi + (size_t)M_DIM * N_DIM;
    _Float16* Bhi = Tlo + (size_t)M_DIM * N_DIM;
    _Float16* Blo = Bhi + (size_t)N_DIM * N_DIM;

    dim3 g1(N_DIM / 256, M_DIM);
    band_conv_split_kernel<<<g1, 256, 0, stream>>>(A, Thi, Tlo);

    dim3 g2(((size_t)N_DIM * N_DIM / 4) / 256);
    split_b_kernel<<<g2, 256, 0, stream>>>(B, Bhi, Blo);

    dim3 g3(N_DIM / BN, M_DIM / BM);
    gemm_tbt_f16x3_kernel<<<g3, 256, 0, stream>>>(Thi, Tlo, Bhi, Blo, out);
}
